// BiLSTM_CRF_60404420051672
// MI455X (gfx1250) — matrix operand order repaired and checked by reading, not compiled, not run on hardware
//
#include <hip/hip_runtime.h>
#include <hip/hip_bf16.h>
#include <math.h>

typedef __attribute__((ext_vector_type(16))) _Float16 v16h;
typedef __attribute__((ext_vector_type(8)))  _Float16 v8h;
typedef __attribute__((ext_vector_type(8)))  float    v8f;

#define BB   32
#define LF   256
#define TT   1024
#define EE   512
#define HH   512
#define G4   2048          // 4*H gate width
#define NT   13
#define DMID 341
#define DPAD 384
#define MROWS (TT*BB)      // 32768 rows, row = t*B + b
#define LDA_T 40           // padded LDS tile row stride (halves): 80B rows -> distinct bank starts

// ---------------- CDNA5 async global->LDS copy (ASYNCcnt path) ----------------
// low 32 bits of a generic pointer into __shared__ are the LDS byte offset (ISA 10.2)
__device__ inline void async_cp16(unsigned lds_off, const _Float16* g) {
    asm volatile("global_load_async_to_lds_b128 %0, %1, off"
                 :: "v"(lds_off), "v"(g) : "memory");
}
__device__ inline void wait_async4() { asm volatile("s_wait_asynccnt 4" ::: "memory"); }
__device__ inline void wait_async0() { asm volatile("s_wait_asynccnt 0" ::: "memory"); }

// ---------------- fragment loaders (CDNA5 16-bit WMMA layouts) ----------------
// A 16x32 (MxK): lane l: m = l&15; elem j: k = (l>>4)*16 + j
__device__ inline v16h frag_a(const _Float16* A, size_t lda, int m0, int k0, int lane) {
    int lo = lane & 15, hi = lane >> 4;
    const _Float16* p = A + (size_t)(m0 + lo) * lda + k0 + hi * 16;
    v8h x0 = *(const v8h*)p;
    v8h x1 = *(const v8h*)(p + 8);
    v16h r;
#pragma unroll
    for (int j = 0; j < 8; ++j) { r[j] = x0[j]; r[j + 8] = x1[j]; }
    return r;
}
// B 32x16 (KxN) where B[k][n] = W[n][k], W row-major N x K:
// lane l: n = l&15; elem j: k = (l>>4)*16 + j  -> contiguous 16 halves of W row
__device__ inline v16h frag_b(const _Float16* W, size_t ldb, int n0, int k0, int lane) {
    int lo = lane & 15, hi = lane >> 4;
    return *(const v16h*)(W + (size_t)(n0 + lo) * ldb + k0 + hi * 16);
}
// same as frag_b but only requires 16B alignment (for padded LDS tiles)
__device__ inline v16h frag_b16(const _Float16* W, size_t ldb, int n0, int k0, int lane) {
    int lo = lane & 15, hi = lane >> 4;
    const _Float16* p = W + (size_t)(n0 + lo) * ldb + k0 + hi * 16;
    v8h x0 = *(const v8h*)p;
    v8h x1 = *(const v8h*)(p + 8);
    v16h r;
#pragma unroll
    for (int j = 0; j < 8; ++j) { r[j] = x0[j]; r[j + 8] = x1[j]; }
    return r;
}

// ---------------- embeddings + layernorm -> fp16 x (T,B,E) ----------------
__global__ __launch_bounds__(256) void k_embed_ln(
    const int* word, const int* chr, const int* pos, const int* spo,
    const float* ew, const float* ec, const float* ep, const float* es,
    const float* g, const float* bta, _Float16* x16)
{
    int row = blockIdx.x;            // row = t*B + b
    int b = row % BB, t = row / BB;
    int seg = t >> 8, tp = t & 255;
    const int* it; const float* tab;
    if      (seg == 0) { it = word; tab = ew; }
    else if (seg == 1) { it = chr;  tab = ec; }
    else if (seg == 2) { it = pos;  tab = ep; }
    else               { it = spo;  tab = es; }
    int idx = it[b * LF + tp];
    const float* e = tab + (size_t)idx * EE;
    __shared__ float red[256];
    int tid = threadIdx.x;
    float v0 = e[tid], v1 = e[tid + 256];
    red[tid] = v0 + v1; __syncthreads();
    for (int s = 128; s > 0; s >>= 1) { if (tid < s) red[tid] += red[tid + s]; __syncthreads(); }
    float mu = red[0] * (1.f / 512.f);
    __syncthreads();
    float d0 = v0 - mu, d1 = v1 - mu;
    red[tid] = d0 * d0 + d1 * d1; __syncthreads();
    for (int s = 128; s > 0; s >>= 1) { if (tid < s) red[tid] += red[tid + s]; __syncthreads(); }
    float rs = rsqrtf(red[0] * (1.f / 512.f) + 1e-5f);
    _Float16* out = x16 + (size_t)row * EE;
    out[tid]       = (_Float16)(d0 * rs * g[tid] + bta[tid]);
    out[tid + 256] = (_Float16)(d1 * rs * g[tid + 256] + bta[tid + 256]);
}

// ---------------- small conversion helpers ----------------
__global__ void k_cvt_f16(const float* s, _Float16* d, int n) {
    int i = blockIdx.x * blockDim.x + threadIdx.x;
    if (i < n) d[i] = (_Float16)s[i];
}
__global__ void k_gate_bias(const float* b, float* out) {     // out[2*G4] = b[d,0,:]+b[d,1,:]
    int i = blockIdx.x * blockDim.x + threadIdx.x;
    if (i < 2 * G4) { int d = i / G4, c = i % G4; out[i] = b[d * 2 * G4 + c] + b[d * 2 * G4 + G4 + c]; }
}
__global__ void k_pad_lin1(const float* w, _Float16* d) {     // (DPAD,1024), zero-pad rows >= DMID
    int i = blockIdx.x * blockDim.x + threadIdx.x;
    if (i < DPAD * 1024) { int n = i / 1024, k = i % 1024; d[i] = (n < DMID) ? (_Float16)w[n * 1024 + k] : (_Float16)0.f; }
}
__global__ void k_pad_bias(const float* b, float* d) {
    int i = blockIdx.x * blockDim.x + threadIdx.x;
    if (i < DPAD) d[i] = (i < DMID) ? b[i] : 0.f;
}

// ---------------- big parallel WMMA GEMM with async double-buffered LDS staging ----------------
// C[M,N] = A[M,K] @ W[N,K]^T + bias. block 256 thr = 8 waves; block tile 128x128;
// per k-step (32) the 128x32 A-tile and 128x32 B-tile are staged via
// GLOBAL_LOAD_ASYNC_TO_LDS_B128 (4 chunks/thread), double-buffered on ASYNCcnt.
__global__ __launch_bounds__(256) void k_gemm(
    const _Float16* __restrict__ A, const _Float16* __restrict__ W, const float* __restrict__ bias,
    _Float16* outH, float* outF, int M, int N, int K)
{
    __shared__ __align__(64) _Float16 ldsA[2][128 * LDA_T];
    __shared__ __align__(64) _Float16 ldsB[2][128 * LDA_T];
    int tid = threadIdx.x;
    int lane = tid & 31, wave = tid >> 5;
    int mw = wave & 3, nw = wave >> 2;
    size_t m0g = (size_t)blockIdx.y * 128;
    size_t n0g = (size_t)blockIdx.x * 128;
    int lo = lane & 15, hi = lane >> 4;

    // staging chunks: c in [0,512): tile row r = c>>2, 16B quarter q = c&3; this thread: c=tid, tid+256
    int r0 = tid >> 2, r1 = 64 + (tid >> 2), q = tid & 3;
    unsigned oA0[2], oA1[2], oB0[2], oB1[2];
#pragma unroll
    for (int s = 0; s < 2; ++s) {
        oA0[s] = (unsigned)(size_t)&ldsA[s][r0 * LDA_T + q * 8];
        oA1[s] = (unsigned)(size_t)&ldsA[s][r1 * LDA_T + q * 8];
        oB0[s] = (unsigned)(size_t)&ldsB[s][r0 * LDA_T + q * 8];
        oB1[s] = (unsigned)(size_t)&ldsB[s][r1 * LDA_T + q * 8];
    }
    const _Float16* pA0 = A + (m0g + r0) * K + q * 8;
    const _Float16* pA1 = A + (m0g + r1) * K + q * 8;
    const _Float16* pB0 = W + (n0g + r0) * K + q * 8;
    const _Float16* pB1 = W + (n0g + r1) * K + q * 8;

    v8f acc[2][4];
#pragma unroll
    for (int mt = 0; mt < 2; ++mt)
#pragma unroll
        for (int nt = 0; nt < 4; ++nt) {
            float bv = bias ? bias[n0g + nw * 64 + nt * 16 + lo] : 0.f;
#pragma unroll
            for (int v = 0; v < 8; ++v) acc[mt][nt][v] = bv;
        }

    // prologue: stage tile 0
    async_cp16(oA0[0], pA0); async_cp16(oA1[0], pA1);
    async_cp16(oB0[0], pB0); async_cp16(oB1[0], pB1);
    pA0 += 32; pA1 += 32; pB0 += 32; pB1 += 32;

    int KT = K / 32;
    for (int kt = 0; kt < KT; ++kt) {
        int cur = kt & 1;
        if (kt + 1 < KT) {
            int nxt = cur ^ 1;
            async_cp16(oA0[nxt], pA0); async_cp16(oA1[nxt], pA1);
            async_cp16(oB0[nxt], pB0); async_cp16(oB1[nxt], pB1);
            pA0 += 32; pA1 += 32; pB0 += 32; pB1 += 32;
            wait_async4();           // tile kt's 4 chunks (issued first, in-order) retired
        } else {
            wait_async0();
        }
        __syncthreads();             // every thread's chunks for tile kt are in LDS
        const _Float16* tA = ldsA[cur];
        const _Float16* tB = ldsB[cur];
        v16h a0 = frag_a(tA, LDA_T, mw * 32,      0, lane);
        v16h a1 = frag_a(tA, LDA_T, mw * 32 + 16, 0, lane);
#pragma unroll
        for (int nt = 0; nt < 4; ++nt) {
            v16h bf = frag_b16(tB, LDA_T, nw * 64 + nt * 16, 0, lane);
            acc[0][nt] = __builtin_amdgcn_wmma_f32_16x16x32_f16(false, a0, false, bf, (short)0, acc[0][nt], false, false);
            acc[1][nt] = __builtin_amdgcn_wmma_f32_16x16x32_f16(false, a1, false, bf, (short)0, acc[1][nt], false, false);
        }
        __syncthreads();             // all reads of buf 'cur' done before it is re-staged
    }
#pragma unroll
    for (int mt = 0; mt < 2; ++mt)
#pragma unroll
        for (int nt = 0; nt < 4; ++nt)
#pragma unroll
            for (int v = 0; v < 8; ++v) {
                size_t row = m0g + mw * 32 + mt * 16 + hi * 8 + v;
                size_t col = n0g + nw * 64 + nt * 16 + lo;
                if (outH) outH[row * (size_t)N + col] = (_Float16)acc[mt][nt][v];
                if (outF) outF[row * (size_t)N + col] = acc[mt][nt][v];
            }
}

// ---------------- persistent LSTM recurrence, one workgroup per direction ----------------
// 1024 thr = 32 waves; wave w owns hidden cols [16w,16w+16): computes all 4 gate tiles
// so the cell update is fully in-register; c state lives in VGPRs across all T steps.
__global__ __launch_bounds__(1024) void k_lstm(
    const _Float16* __restrict__ xg,   // (2,T,B,G4) fp16 gate preacts (both biases folded)
    const _Float16* __restrict__ whh,  // (2,G4,HH) fp16
    _Float16* __restrict__ hout)       // (T,B,2H) fp16, dir offset in feature dim
{
    int dir = blockIdx.x;
    const _Float16* xgd = xg  + (size_t)dir * MROWS * G4;
    const _Float16* wd  = whh + (size_t)dir * G4 * HH;
    int dOff = dir * HH;
    __shared__ __align__(32) _Float16 hsh[BB * HH];
    int tid = threadIdx.x;
    int lane = tid & 31, wave = tid >> 5;
    int lo = lane & 15, hi = lane >> 4;
    int j0 = wave * 16;
    for (int i = tid; i < BB * HH; i += 1024) hsh[i] = (_Float16)0.f;
    float c[2][8];
#pragma unroll
    for (int mt = 0; mt < 2; ++mt)
#pragma unroll
        for (int v = 0; v < 8; ++v) c[mt][v] = 0.f;
    __syncthreads();
    for (int step = 0; step < TT; ++step) {
        int t = dir ? (TT - 1 - step) : step;
        // hide next timestep's 128KB xg row behind this step's WMMA chain
        if (step + 1 < TT) {
            int tn = dir ? (TT - 2 - step) : (step + 1);
            __builtin_prefetch(xgd + (size_t)tn * BB * G4 + (size_t)tid * 64, 0, 0);
        }
        const _Float16* xrow = xgd + (size_t)t * BB * G4;
        v8f acc[2][4];
#pragma unroll
        for (int mt = 0; mt < 2; ++mt)
#pragma unroll
            for (int g = 0; g < 4; ++g)
#pragma unroll
                for (int v = 0; v < 8; ++v)
                    acc[mt][g][v] = (float)xrow[(size_t)(mt * 16 + hi * 8 + v) * G4 + g * HH + j0 + lo];
        for (int k0 = 0; k0 < HH; k0 += 32) {
            v16h a0 = frag_a(hsh, HH, 0,  k0, lane);
            v16h a1 = frag_a(hsh, HH, 16, k0, lane);
#pragma unroll
            for (int g = 0; g < 4; ++g) {
                v16h bf = frag_b(wd, HH, g * HH + j0, k0, lane);
                acc[0][g] = __builtin_amdgcn_wmma_f32_16x16x32_f16(false, a0, false, bf, (short)0, acc[0][g], false, false);
                acc[1][g] = __builtin_amdgcn_wmma_f32_16x16x32_f16(false, a1, false, bf, (short)0, acc[1][g], false, false);
            }
        }
        float hval[2][8];
#pragma unroll
        for (int mt = 0; mt < 2; ++mt)
#pragma unroll
            for (int v = 0; v < 8; ++v) {
                float ig = 1.f / (1.f + expf(-acc[mt][0][v]));
                float fg = 1.f / (1.f + expf(-acc[mt][1][v]));
                float gt = tanhf(acc[mt][2][v]);
                float og = 1.f / (1.f + expf(-acc[mt][3][v]));
                float cv = fg * c[mt][v] + ig * gt;
                c[mt][v] = cv;
                hval[mt][v] = og * tanhf(cv);
            }
        __syncthreads();   // everyone done reading previous h
#pragma unroll
        for (int mt = 0; mt < 2; ++mt)
#pragma unroll
            for (int v = 0; v < 8; ++v) {
                int brow = mt * 16 + hi * 8 + v;
                _Float16 hh = (_Float16)hval[mt][v];
                hsh[brow * HH + j0 + lo] = hh;
                hout[((size_t)t * BB + brow) * (2 * HH) + dOff + j0 + lo] = hh;
            }
        __syncthreads();   // new h visible
    }
}

// ---------------- per-row LN + leaky-relu + lin2 -> logits (row = t*B+b) ----------------
__global__ __launch_bounds__(128) void k_head(
    const float* __restrict__ mid, const float* g2, const float* b2,
    const float* wlin2, const float* blin2, float* logits)
{
    int row = blockIdx.x;
    const float* m = mid + (size_t)row * DPAD;
    __shared__ float xn[DMID];
    __shared__ float red[128];
    int tid = threadIdx.x;
    float s = 0.f;
    for (int k = tid; k < DMID; k += 128) s += m[k];
    red[tid] = s; __syncthreads();
    for (int st = 64; st > 0; st >>= 1) { if (tid < st) red[tid] += red[tid + st]; __syncthreads(); }
    float mu = red[0] / (float)DMID;
    __syncthreads();
    float s2 = 0.f;
    for (int k = tid; k < DMID; k += 128) { float d = m[k] - mu; s2 += d * d; }
    red[tid] = s2; __syncthreads();
    for (int st = 64; st > 0; st >>= 1) { if (tid < st) red[tid] += red[tid + st]; __syncthreads(); }
    float rs = rsqrtf(red[0] / (float)DMID + 1e-5f);
    for (int k = tid; k < DMID; k += 128) {
        float v = (m[k] - mu) * rs * g2[k] + b2[k];
        xn[k] = v > 0.f ? v : 0.01f * v;
    }
    __syncthreads();
    if (tid < NT) {
        float a = blin2[tid];
        const float* wr = wlin2 + tid * DMID;
        for (int k = 0; k < DMID; ++k) a += wr[k] * xn[k];
        logits[(size_t)row * NT + tid] = a;
    }
}

// ---------------- CRF forward scan + gold score + mean NLL -> scalar ----------------
__global__ __launch_bounds__(512) void k_crf(
    const float* __restrict__ logits, const int* __restrict__ target,
    const float* __restrict__ trans, float* out)
{
    __shared__ float alpha[BB * NT];
    __shared__ float tr[NT * NT];
    __shared__ float red[512];
    __shared__ float normv[BB];
    int tid = threadIdx.x;
    if (tid < NT * NT) tr[tid] = trans[tid];
    if (tid < BB * NT) {
        int b = tid / NT, j = tid % NT;
        alpha[tid] = logits[(size_t)b * NT + j];          // t = 0
    }
    __syncthreads();
    for (int t = 1; t < TT; ++t) {
        float val = 0.f;
        if (tid < BB * NT) {
            int b = tid / NT, j = tid % NT;
            float mx = -1e30f;
#pragma unroll
            for (int i = 0; i < NT; ++i) { float v = alpha[b * NT + i] + tr[i * NT + j]; mx = v > mx ? v : mx; }
            float sm = 0.f;
#pragma unroll
            for (int i = 0; i < NT; ++i) sm += expf(alpha[b * NT + i] + tr[i * NT + j] - mx);
            val = mx + logf(sm) + logits[(size_t)(t * BB + b) * NT + j];
        }
        __syncthreads();
        if (tid < BB * NT) alpha[tid] = val;
        __syncthreads();
    }
    if (tid < BB) {
        float mx = -1e30f;
        for (int j = 0; j < NT; ++j) { float v = alpha[tid * NT + j]; mx = v > mx ? v : mx; }
        float sm = 0.f;
        for (int j = 0; j < NT; ++j) sm += expf(alpha[tid * NT + j] - mx);
        normv[tid] = mx + logf(sm);
    }
    { // gold partials: 32 b x 16 slices
        int b = tid >> 4, sl = tid & 15;
        float acc = 0.f;
        for (int t = sl; t < TT; t += 16) {
            int tg = target[b * TT + t];
            acc += logits[(size_t)(t * BB + b) * NT + tg];
            if (t > 0) { int tp = target[b * TT + t - 1]; acc += tr[tp * NT + tg]; }
        }
        red[tid] = acc;
    }
    __syncthreads();
    if (tid < BB) {
        float gold = 0.f;
        for (int s = 0; s < 16; ++s) gold += red[tid * 16 + s];
        red[tid] = normv[tid] - gold;
    }
    __syncthreads();
    if (tid == 0) {
        float s = 0.f;
        for (int b = 0; b < BB; ++b) s += red[b];
        out[0] = s / (float)BB;
    }
}

// ---------------- host launcher ----------------
extern "C" void kernel_launch(void* const* d_in, const int* in_sizes, int n_in,
                              void* d_out, int out_size, void* d_ws, size_t ws_size,
                              hipStream_t stream)
{
    (void)in_sizes; (void)n_in; (void)out_size; (void)ws_size;
    const int*   word  = (const int*)d_in[0];
    const int*   chr   = (const int*)d_in[1];
    const int*   pos   = (const int*)d_in[2];
    const int*   spo   = (const int*)d_in[3];
    const int*   tgt   = (const int*)d_in[4];
    const float* ew    = (const float*)d_in[5];
    const float* ec    = (const float*)d_in[6];
    const float* ep    = (const float*)d_in[7];
    const float* es    = (const float*)d_in[8];
    const float* ln1g  = (const float*)d_in[9];
    const float* ln1b  = (const float*)d_in[10];
    const float* Wih0  = (const float*)d_in[11];
    const float* Whh0  = (const float*)d_in[12];
    const float* b0    = (const float*)d_in[13];
    const float* Wih1  = (const float*)d_in[14];
    const float* Whh1  = (const float*)d_in[15];
    const float* b1    = (const float*)d_in[16];
    const float* Wlin1 = (const float*)d_in[17];
    const float* blin1 = (const float*)d_in[18];
    const float* ln2g  = (const float*)d_in[19];
    const float* ln2b  = (const float*)d_in[20];
    const float* Wlin2 = (const float*)d_in[21];
    const float* blin2 = (const float*)d_in[22];
    const float* trans = (const float*)d_in[23];

    char* ws = (char*)d_ws;
    size_t off = 0;
    auto alloc = [&](size_t bytes) { void* p = ws + off; off = (off + bytes + 255) & ~(size_t)255; return p; };

    _Float16* x16     = (_Float16*)alloc((size_t)MROWS * EE * 2);
    _Float16* h0      = (_Float16*)alloc((size_t)MROWS * 2 * HH * 2);
    _Float16* h1      = (_Float16*)alloc((size_t)MROWS * 2 * HH * 2);
    _Float16* wih0_16 = (_Float16*)alloc((size_t)2 * G4 * EE * 2);
    _Float16* whh0_16 = (_Float16*)alloc((size_t)2 * G4 * HH * 2);
    _Float16* wih1_16 = (_Float16*)alloc((size_t)2 * G4 * 1024 * 2);
    _Float16* whh1_16 = (_Float16*)alloc((size_t)2 * G4 * HH * 2);
    _Float16* wlin116 = (_Float16*)alloc((size_t)DPAD * 1024 * 2);
    float*    bias0   = (float*)alloc((size_t)2 * G4 * 4);
    float*    bias1   = (float*)alloc((size_t)2 * G4 * 4);
    float*    blin1p  = (float*)alloc((size_t)DPAD * 4);
    _Float16* xg      = (_Float16*)alloc((size_t)2 * MROWS * G4 * 2);
    float*    mid     = (float*)alloc((size_t)MROWS * DPAD * 4);
    float*    logits  = (float*)alloc((size_t)MROWS * NT * 4);

    // 1) embeddings + LN -> fp16 (T,B,E)
    k_embed_ln<<<MROWS, 256, 0, stream>>>(word, chr, pos, spo, ew, ec, ep, es, ln1g, ln1b, x16);

    // 2) weight conversions / bias folds
    int n;
    n = 2 * G4 * EE;   k_cvt_f16<<<(n + 255) / 256, 256, 0, stream>>>(Wih0, wih0_16, n);
    n = 2 * G4 * HH;   k_cvt_f16<<<(n + 255) / 256, 256, 0, stream>>>(Whh0, whh0_16, n);
    n = 2 * G4 * 1024; k_cvt_f16<<<(n + 255) / 256, 256, 0, stream>>>(Wih1, wih1_16, n);
    n = 2 * G4 * HH;   k_cvt_f16<<<(n + 255) / 256, 256, 0, stream>>>(Whh1, whh1_16, n);
    n = 2 * G4;        k_gate_bias<<<(n + 255) / 256, 256, 0, stream>>>(b0, bias0);
                       k_gate_bias<<<(n + 255) / 256, 256, 0, stream>>>(b1, bias1);
    n = DPAD * 1024;   k_pad_lin1<<<(n + 255) / 256, 256, 0, stream>>>(Wlin1, wlin116);
    k_pad_bias<<<2, 256, 0, stream>>>(blin1, blin1p);

    // 3) layer0 input GEMMs (off critical path): xg[d] = x @ Wih0[d]^T + (bih+bhh)
    for (int d = 0; d < 2; ++d)
        k_gemm<<<dim3(G4 / 128, MROWS / 128), 256, 0, stream>>>(
            x16, wih0_16 + (size_t)d * G4 * EE, bias0 + (size_t)d * G4,
            xg + (size_t)d * MROWS * G4, nullptr, MROWS, G4, EE);

    // 4) layer0 recurrence: 2 persistent workgroups (fwd/bwd)
    k_lstm<<<2, 1024, 0, stream>>>(xg, whh0_16, h0);

    // 5) layer1 input GEMMs (K = 2H = 1024)
    for (int d = 0; d < 2; ++d)
        k_gemm<<<dim3(G4 / 128, MROWS / 128), 256, 0, stream>>>(
            h0, wih1_16 + (size_t)d * G4 * 1024, bias1 + (size_t)d * G4,
            xg + (size_t)d * MROWS * G4, nullptr, MROWS, G4, 1024);

    // 6) layer1 recurrence
    k_lstm<<<2, 1024, 0, stream>>>(xg, whh1_16, h1);

    // 7) lin1 GEMM: mid = h1 @ Wlin1^T + b (N padded 341 -> 384)
    k_gemm<<<dim3(DPAD / 128, MROWS / 128), 256, 0, stream>>>(
        h1, wlin116, blin1p, nullptr, mid, MROWS, DPAD, 1024);

    // 8) per-row LN + leaky + lin2 -> logits
    k_head<<<MROWS, 128, 0, stream>>>(mid, ln2g, ln2b, Wlin2, blin2, logits);

    // 9) CRF NLL -> scalar
    k_crf<<<1, 512, 0, stream>>>(logits, tgt, trans, (float*)d_out);
}
